// AutoCorrelation_43104291782775
// MI455X (gfx1250) — compile-verified, matrix operand order checked by reading, not run on hardware
//
#include <hip/hip_runtime.h>
#include <hip/hip_bf16.h>
#include <math.h>

// Problem constants (B, L, D_MODEL, N_HEADS from reference)
#define B_   4
#define L_   2048
#define DM   512
#define NH   8
#define TOPK 7   // max(1, int(1*log(2048))) = int(7.624) = 7

// Padded LDS row stride (ushorts): 40 -> 80B rows -> fragment loads hit 16 distinct banks
#define LDSTRIDE 40

typedef __attribute__((ext_vector_type(16))) __bf16 v16bf;
typedef __attribute__((ext_vector_type(8)))  float  v8f;
typedef __attribute__((ext_vector_type(2)))  float  v2f;

__device__ __forceinline__ unsigned short f32_to_bf16_rne(float f) {
  unsigned int x = __float_as_uint(f);
  unsigned int r = x + 0x7FFFu + ((x >> 16) & 1u);
  return (unsigned short)(r >> 16);
}

// Async global->LDS 16B copy (tracked by ASYNCcnt).
// LDS generic pointers carry the LDS byte offset in addr[31:0] (aperture rule).
__device__ __forceinline__ void async_copy_b128(unsigned short* lds_dst,
                                                const unsigned short* gsrc) {
  unsigned lds_off = (unsigned)(uintptr_t)lds_dst;
  unsigned long long ga = (unsigned long long)(uintptr_t)gsrc;
  asm volatile("global_load_async_to_lds_b128 %0, %1, off"
               :: "v"(lds_off), "v"(ga)
               : "memory");
}
__device__ __forceinline__ void wait_asynccnt0() {
  asm volatile("s_wait_asynccnt 0x0" ::: "memory");
}

// ---------------------------------------------------------------- k1: qmean[b,d] = mean_l queries[b,l,d]
__global__ void k1_qmean(const float* __restrict__ q, float* __restrict__ qmean) {
  const int b = blockIdx.x;
  const int d = threadIdx.x;            // 512 threads
  const float* qb = q + (size_t)b * L_ * DM + d;
  float s = 0.f;
  for (int l = 0; l < L_; ++l) s += qb[(size_t)l * DM];
  qmean[b * DM + d] = s * (1.0f / L_);
}

// ---------------------------------------------------------------- k2: m[b,:] = (1/H) * wk * (qmean @ wq)
__global__ void k2_mvec(const float* __restrict__ qmean, const float* __restrict__ wq,
                        const float* __restrict__ wk, float* __restrict__ mvec) {
  __shared__ float t1[DM];
  const int b = blockIdx.x;
  const int d = threadIdx.x;            // 512 threads
  float s = 0.f;
  for (int e = 0; e < DM; ++e) s += qmean[b * DM + e] * wq[e * DM + d];
  t1[d] = s;
  __syncthreads();
  float m = 0.f;
  for (int e = 0; e < DM; ++e) m += wk[d * DM + e] * t1[e];
  mvec[b * DM + d] = m * (1.0f / NH);
}

// ---------------------------------------------------------------- k3: mean_corr[b,j] = keys[b,j,:] . m[b,:]
__global__ void __launch_bounds__(256) k3_mcorr(const float* __restrict__ keys,
                                                const float* __restrict__ mvec,
                                                float* __restrict__ mc) {
  const int lane = threadIdx.x & 31;
  const int gw   = blockIdx.x * 8 + (threadIdx.x >> 5);   // 0..B*L-1, one wave per row
  const int b = gw >> 11;                                 // / 2048
  const int j = gw & (L_ - 1);
  const float* kr = keys + ((size_t)b * L_ + j) * DM;
  const float* mb = mvec + b * DM;
  float s = 0.f;
  for (int i = 0; i < DM / 32; ++i) { const int e = lane + 32 * i; s += kr[e] * mb[e]; }
  for (int off = 16; off; off >>= 1) s += __shfl_xor(s, off, 32);
  if (lane == 0) mc[gw] = s;
}

// ---------------------------------------------------------------- k4: top-7 + softmax per batch (one wave)
__global__ void k4_topk(const float* __restrict__ mc, float* __restrict__ wts,
                        int* __restrict__ dls) {
  const int b = blockIdx.x;
  const int lane = threadIdx.x;         // 32 threads
  __shared__ int   seli[TOPK];
  __shared__ float selv[TOPK];
  const float* m = mc + b * L_;
  for (int r = 0; r < TOPK; ++r) {
    float bv = -3.4e38f; int bi = 0;
    for (int j = lane; j < L_; j += 32) {
      bool used = false;
      for (int p = 0; p < r; ++p) used |= (seli[p] == j);
      const float v = m[j];
      if (!used && v > bv) { bv = v; bi = j; }
    }
    for (int off = 16; off; off >>= 1) {
      const float ov = __shfl_xor(bv, off, 32);
      const int   oi = __shfl_xor(bi, off, 32);
      if (ov > bv || (ov == bv && oi < bi)) { bv = ov; bi = oi; }
    }
    if (lane == 0) { seli[r] = bi; selv[r] = bv; }
    __syncthreads();
  }
  if (lane == 0) {
    float mx = selv[0];
    for (int p = 1; p < TOPK; ++p) mx = fmaxf(mx, selv[p]);
    float ex[TOPK], s = 0.f;
    for (int p = 0; p < TOPK; ++p) { ex[p] = __expf(selv[p] - mx); s += ex[p]; }
    for (int p = 0; p < TOPK; ++p) { wts[b * 8 + p] = ex[p] / s; dls[b * 8 + p] = seli[p]; }
  }
}

// ---------------------------------------------------------------- k5: wvoT = (wv @ wo)^T  (exact fp32 WMMA 16x16x4)
// Stored N-major (wvoT[n][k]) so the big GEMM's B-fragments are K-contiguous in LDS,
// and each lane's 8-output epilogue is one contiguous global_store_b128.
__global__ void __launch_bounds__(256) k5_wvo(const float* __restrict__ wv,
                                              const float* __restrict__ wo,
                                              unsigned short* __restrict__ wvoT) {
  const int lane = threadIdx.x & 31;
  const int wave = threadIdx.x >> 5;
  const int tile = blockIdx.x * 8 + wave;   // 0..1023 (32x32 tiles of 16x16)
  const int tm = (tile >> 5) * 16;
  const int tn = (tile & 31) * 16;
  const int r16  = lane & 15;
  const int koff = (lane < 16) ? 0 : 2;     // 32-bit A 16x4 layout: lanes 16-31 hold K=2,3
  v8f acc = {};
  for (int k = 0; k < DM; k += 4) {
    v2f a, b;
    a.x = wv[(size_t)(tm + r16) * DM + k + koff];
    a.y = wv[(size_t)(tm + r16) * DM + k + koff + 1];
    b.x = wo[(size_t)(k + koff) * DM + tn + r16];
    b.y = wo[(size_t)(k + koff + 1) * DM + tn + r16];
    acc = __builtin_amdgcn_wmma_f32_16x16x4_f32(false, a, false, b, (short)0, acc,
                                                false, false);
  }
  const int half = (lane < 16) ? 0 : 8;     // C/D layout: lanes 16-31 -> M = r+8
  for (int r = 0; r < 8; ++r)
    wvoT[(size_t)(tn + r16) * DM + tm + r + half] = f32_to_bf16_rne(acc[r]);
}

// ---------------------------------------------------------------- k5b: aggv[b,t,:] = bf16( sum_k w_k * values[b,(t+d_k)%L,:] )
__global__ void __launch_bounds__(256) k5b_aggv(const float* __restrict__ values,
                                                const float* __restrict__ wts,
                                                const int* __restrict__ dls,
                                                unsigned short* __restrict__ aggv) {
  const int b    = blockIdx.y;
  const int lane = threadIdx.x & 31;
  const int rloc = threadIdx.x >> 5;            // 0..7
  const int t    = blockIdx.x * 8 + rloc;
  float w[TOPK]; int rw[TOPK];
#pragma unroll
  for (int k = 0; k < TOPK; ++k) {
    w[k]  = wts[b * 8 + k];
    rw[k] = (t + dls[b * 8 + k]) & (L_ - 1);    // circular roll
  }
  const float* vb = values + (size_t)b * L_ * DM;
  unsigned short* ob = aggv + ((size_t)b * L_ + t) * DM;
  for (int i = 0; i < DM / 32; ++i) {
    const int e = lane + i * 32;                // coalesced within wave
    float acc = 0.f;
#pragma unroll
    for (int k = 0; k < TOPK; ++k) acc += w[k] * vb[(size_t)rw[k] * DM + e];
    ob[e] = f32_to_bf16_rne(acc);
  }
}

// ---------------------------------------------------------------- k6: out = aggv(bf16) @ wvoT(bf16)^T, f32 accum
// Block tile 128(M) x 64(N), K-step 32, 8 waves: wave w -> M rows [w*16, w*16+16), 4 accumulators.
// Double-buffered async staging: global_load_async_to_lds_b128 fills tile t+1 while the
// WMMAs consume tile t. One s_barrier per K-step. Fragments are bank-conflict-free
// ds_load_b128 thanks to the 40-ushort padded row stride.
__global__ void __launch_bounds__(256) k6_gemm(const unsigned short* __restrict__ Abf,
                                               const unsigned short* __restrict__ BbfT,
                                               float* __restrict__ out) {
  __shared__ __align__(16) unsigned short As[2][128 * LDSTRIDE];   // [m][k] padded
  __shared__ __align__(16) unsigned short BsT[2][64 * LDSTRIDE];   // [n][k] padded
  const int tid  = threadIdx.x;
  const int lane = tid & 31;
  const int wave = tid >> 5;
  const int m0 = blockIdx.x * 128;
  const int n0 = blockIdx.y * 64;
  const int mrow = lane & 15;
  const int half = (lane < 16) ? 0 : 8;   // 16-bit layout: lanes 0-15 hold K 0-7/16-23, lanes 16-31 hold K 8-15/24-31

  const int ar = tid >> 1, ag = (tid & 1) * 16;   // A-tile load coords (ushort units)
  const int br = tid >> 2, bg = (tid & 3) * 8;    // B-tile load coords

  const unsigned short* gA = Abf + (size_t)(m0 + ar) * DM + ag;
  const unsigned short* gB = BbfT + (size_t)(n0 + br) * DM + bg;

  v8f acc[4] = {};

  // Prologue: stage K-tile 0 into buffer 0
  async_copy_b128(&As[0][ar * LDSTRIDE + ag], gA);
  async_copy_b128(&As[0][ar * LDSTRIDE + ag + 8], gA + 8);
  async_copy_b128(&BsT[0][br * LDSTRIDE + bg], gB);

  for (int it = 0; it < DM / 32; ++it) {
    const int cur = it & 1;
    wait_asynccnt0();        // my async writes into buffer `cur` are done
    __syncthreads();         // everyone's writes done; everyone's reads of buffer cur^1 done

    // Kick off DMA for the next K-tile into the alternate buffer while we compute
    if (it + 1 < DM / 32) {
      const int koff2 = (it + 1) * 32;
      async_copy_b128(&As[cur ^ 1][ar * LDSTRIDE + ag], gA + koff2);
      async_copy_b128(&As[cur ^ 1][ar * LDSTRIDE + ag + 8], gA + koff2 + 8);
      async_copy_b128(&BsT[cur ^ 1][br * LDSTRIDE + bg], gB + koff2);
    }

    // A fragment (per ISA 16-bit A 16x32 table) — two conflict-free ds_load_b128
    union { v16bf v; uint4 q[2]; } af;
    const int arow = wave * 16 + mrow;
    af.q[0] = *reinterpret_cast<const uint4*>(&As[cur][arow * LDSTRIDE + half]);       // K = half..half+7
    af.q[1] = *reinterpret_cast<const uint4*>(&As[cur][arow * LDSTRIDE + half + 16]);  // K = half+16..half+23

#pragma unroll
    for (int j = 0; j < 4; ++j) {
      // B fragment: lane -> column n = lane%16, K packing mirrors A — two ds_load_b128
      union { v16bf v; uint4 q[2]; } bfg;
      const int nrow = j * 16 + mrow;
      bfg.q[0] = *reinterpret_cast<const uint4*>(&BsT[cur][nrow * LDSTRIDE + half]);
      bfg.q[1] = *reinterpret_cast<const uint4*>(&BsT[cur][nrow * LDSTRIDE + half + 16]);
      acc[j] = __builtin_amdgcn_wmma_f32_16x16x32_bf16(false, af.v, false, bfg.v,
                                                       (short)0, acc[j], false, false);
    }
  }

  // C/D f32 16x16 layout: lane -> N=lane%16; VGPR r -> M = r + (lane<16 ? 0 : 8)
#pragma unroll
  for (int j = 0; j < 4; ++j) {
    const int gcol = n0 + j * 16 + mrow;
#pragma unroll
    for (int r = 0; r < 8; ++r) {
      const int grow = m0 + wave * 16 + r + half;
      out[(size_t)grow * DM + gcol] = acc[j][r];
    }
  }
}

// ----------------------------------------------------------------
extern "C" void kernel_launch(void* const* d_in, const int* in_sizes, int n_in,
                              void* d_out, int out_size, void* d_ws, size_t ws_size,
                              hipStream_t stream) {
  const float* q  = (const float*)d_in[0];
  const float* ke = (const float*)d_in[1];
  const float* va = (const float*)d_in[2];
  const float* wq = (const float*)d_in[3];
  const float* wk = (const float*)d_in[4];
  const float* wv = (const float*)d_in[5];
  const float* wo = (const float*)d_in[6];
  float* out = (float*)d_out;

  char* ws = (char*)d_ws;
  float*          qmean = (float*)(ws + 0);        //  4*512 f32
  float*          mvec  = (float*)(ws + 8192);     //  4*512 f32
  float*          mc    = (float*)(ws + 16384);    //  4*2048 f32
  float*          wts   = (float*)(ws + 49152);    //  4*8 f32
  int*            dls   = (int*)  (ws + 49280);    //  4*8 i32
  unsigned short* wvoT  = (unsigned short*)(ws + 65536);    // 512*512 bf16 (N-major)
  unsigned short* aggv  = (unsigned short*)(ws + 589824);   // 8192*512 bf16

  k1_qmean<<<B_, DM, 0, stream>>>(q, qmean);
  k2_mvec <<<B_, DM, 0, stream>>>(qmean, wq, wk, mvec);
  k3_mcorr<<<(B_ * L_) / 8, 256, 0, stream>>>(ke, mvec, mc);
  k4_topk <<<B_, 32, 0, stream>>>(mc, wts, dls);
  k5_wvo  <<<(32 * 32) / 8, 256, 0, stream>>>(wv, wo, wvoT);
  k5b_aggv<<<dim3(L_ / 8, B_), 256, 0, stream>>>(va, wts, dls, aggv);
  k6_gemm <<<dim3((B_ * L_) / 128, DM / 64), 256, 0, stream>>>(aggv, wvoT, out);
}
